// BipartiteGAT_40072044871746
// MI455X (gfx1250) — matrix operand order repaired and checked by reading, not compiled, not run on hardware
//
#include <hip/hip_runtime.h>
#include <hip/hip_bf16.h>
#include <math.h>

// ---------------- problem constants (match reference) ----------------
constexpr int NS = 50000, ND = 50000, EE = 300000;
constexpr int SD = 64, RD = 32, HID = 64, HEADS = 4, ED = 16;
constexpr int HO = HEADS * HID;            // 256
constexpr float SLOPE = 0.2f;
constexpr float LN_EPS = 1e-5f;

typedef __attribute__((ext_vector_type(16))) __bf16 v16bf;
typedef __attribute__((ext_vector_type(8)))  float  v8f;

// ---------------- helpers ----------------
__device__ __forceinline__ unsigned enc_ord(float f) {
    unsigned b = __float_as_uint(f);
    return (b & 0x80000000u) ? ~b : (b | 0x80000000u);
}
__device__ __forceinline__ float dec_ord(unsigned u) {
    unsigned b = (u & 0x80000000u) ? (u ^ 0x80000000u) : ~u;
    float f = __uint_as_float(b);
    return isfinite(f) ? f : 0.0f;   // mirrors reference where(isfinite(m), m, 0)
}

// ---------------- fp32 -> bf16 cast ----------------
__global__ void cast_bf16(const float* __restrict__ in, __bf16* __restrict__ out, int n) {
    for (int i = blockIdx.x * blockDim.x + threadIdx.x; i < n; i += gridDim.x * blockDim.x)
        out[i] = (__bf16)in[i];
}

// ---------------- WMMA GEMM:  Y[N, NCOLS] = X[N,K] @ W[NCOLS,K]^T (+bias) ----------------
// One wave computes one 16x16 output tile with v_wmma_f32_16x16x32_bf16.
// Block = 128 threads = 4 waves; gridDim.y * 64 must equal NCOLS.
template<int K, int NCOLS, bool BIAS, typename OutT>
__global__ void gemm_wmma(const __bf16* __restrict__ X, const __bf16* __restrict__ W,
                          const float* __restrict__ bias, OutT* __restrict__ Y) {
    const int lane = threadIdx.x & 31;
    const int wave = threadIdx.x >> 5;
    const int r    = lane & 15;     // A row for loads; D column for stores
    const int hi   = lane >> 4;
    const int rowBase = blockIdx.x * 16;
    const int colBase = (blockIdx.y * 4 + wave) * 16;

    v8f acc = {};
#pragma unroll
    for (int k0 = 0; k0 < K; k0 += 32) {
        v16bf a, b;
        const __bf16* Xr = X + (size_t)(rowBase + r) * K + k0;
#pragma unroll
        for (int i = 0; i < 16; ++i) {
            // 16-bit A layout: lanes 0-15 hold K {0..7,16..23}, lanes 16-31 {8..15,24..31}
            int kk = (i < 8) ? (hi * 8 + i) : (16 + hi * 8 + (i - 8));
            a[i] = (k0 + kk < K) ? Xr[kk] : (__bf16)0.0f;
        }
        const __bf16* Wr = W + (size_t)(colBase + r) * K + k0;
#pragma unroll
        for (int i = 0; i < 16; ++i) {
            // B layout: lanes 0-15 K {0..7,16..23}, lanes 16-31 {8..15,24..31} (column = lane&15)
            int kk = (i < 8) ? (hi * 8 + i) : (16 + hi * 8 + (i - 8));
            b[i] = (k0 + kk < K) ? Wr[kk] : (__bf16)0.0f;
        }
        acc = __builtin_amdgcn_wmma_f32_16x16x32_bf16(false, a, false, b,
                                                      (short)0, acc, false, false);
    }
    const int n = colBase + r;
    float bv = BIAS ? bias[n] : 0.0f;
#pragma unroll
    for (int j = 0; j < 8; ++j) {
        int m = rowBase + hi * 8 + j;     // D layout: lanes 16-31 hold M=8..15
        Y[(size_t)m * NCOLS + n] = (OutT)(acc[j] + bv);
    }
}

// ---------------- edge pass A: logits + segment max ----------------
// One wave per edge; lane handles 8 contiguous channels; head = lane>>3.
__global__ void edge_logits(const __bf16* __restrict__ xl, const __bf16* __restrict__ xr,
                            const __bf16* __restrict__ xe, const int* __restrict__ ei,
                            const float* __restrict__ att, float* __restrict__ logits,
                            unsigned* __restrict__ menc) {
    const int wave = threadIdx.x >> 5, lane = threadIdx.x & 31;
    const long e = (long)blockIdx.x * 8 + wave;
    const int src = ei[e], dst = ei[(long)EE + e];
    const int base = lane * 8, h = lane >> 3;
    const __bf16* pj = xl + (size_t)src * HO + base;
    const __bf16* pr = xr + (size_t)dst * HO + base;
    const __bf16* pe = xe + (size_t)e  * HO + base;
    const float*  pa = att + base;                 // att flat [HEADS*HID], col index == base+k
    float part = 0.0f;
#pragma unroll
    for (int k = 0; k < 8; ++k) {
        float v = (float)pj[k] + (float)pr[k] + (float)pe[k];
        v = (v > 0.0f) ? v : SLOPE * v;            // GATv2 leaky_relu before attention
        part += v * pa[k];
    }
    part += __shfl_xor(part, 1, 32);
    part += __shfl_xor(part, 2, 32);
    part += __shfl_xor(part, 4, 32);
    if ((lane & 7) == 0) {
        logits[e * HEADS + h] = part;
        atomicMax(&menc[(size_t)dst * HEADS + h], enc_ord(part));
    }
}

// ---------------- edge pass B: exp + segment denom ----------------
__global__ void edge_exp(float* __restrict__ logits, const unsigned* __restrict__ menc,
                         float* __restrict__ denom, const int* __restrict__ ei) {
    const long t = (long)blockIdx.x * blockDim.x + threadIdx.x;
    if (t >= (long)EE * HEADS) return;
    const long e = t >> 2; const int h = (int)(t & 3);
    const int dst = ei[(long)EE + e];
    const float m = dec_ord(menc[(size_t)dst * HEADS + h]);
    const float ex = expf(logits[t] - m);
    logits[t] = ex;                                 // reuse buffer as exp(logit - m)
    unsafeAtomicAdd(&denom[(size_t)dst * HEADS + h], ex);
}

// ---------------- edge pass C: alpha-weighted scatter of messages ----------------
__global__ void edge_scatter(const __bf16* __restrict__ xl, const float* __restrict__ exbuf,
                             const float* __restrict__ denom, const int* __restrict__ ei,
                             float* __restrict__ segout) {
    const int wave = threadIdx.x >> 5, lane = threadIdx.x & 31;
    const long e = (long)blockIdx.x * 8 + wave;
    const int src = ei[e], dst = ei[(long)EE + e];
    const int base = lane * 8, h = lane >> 3;
    const float alpha = exbuf[e * HEADS + h] /
                        (denom[(size_t)dst * HEADS + h] + 1e-16f);
    const __bf16* pj = xl + (size_t)src * HO + base;
    float* po = segout + (size_t)dst * HO + base;
#pragma unroll
    for (int k = 0; k < 8; ++k)
        unsafeAtomicAdd(&po[k], (float)pj[k] * alpha);
}

// ---------------- per-node: head-mean + bias + residual + LayerNorm ----------------
// One wave per node; 64 channels -> 2 per lane; wave-wide shuffle reductions.
__global__ void node_ln(const float* __restrict__ segout, const float* __restrict__ bias,
                        const float* __restrict__ res, const float* __restrict__ g,
                        const float* __restrict__ be, float* __restrict__ xout,
                        __bf16* __restrict__ xbf, int writebf) {
    const int wave = threadIdx.x >> 5, lane = threadIdx.x & 31;
    const long d = (long)blockIdx.x * 8 + wave;
    const int i0 = lane, i1 = lane + 32;
    const float* sg = segout + (size_t)d * HO;
    float v0 = 0.25f * (sg[i0] + sg[64 + i0] + sg[128 + i0] + sg[192 + i0])
             + bias[i0] + res[(size_t)d * HID + i0];
    float v1 = 0.25f * (sg[i1] + sg[64 + i1] + sg[128 + i1] + sg[192 + i1])
             + bias[i1] + res[(size_t)d * HID + i1];
    float s = v0 + v1, sq = v0 * v0 + v1 * v1;
#pragma unroll
    for (int off = 1; off < 32; off <<= 1) {
        s  += __shfl_xor(s,  off, 32);
        sq += __shfl_xor(sq, off, 32);
    }
    const float mean = s * (1.0f / HID);
    const float var  = sq * (1.0f / HID) - mean * mean;
    const float rs   = rsqrtf(var + LN_EPS);
    const float y0 = (v0 - mean) * rs * g[i0] + be[i0];
    const float y1 = (v1 - mean) * rs * g[i1] + be[i1];
    xout[(size_t)d * HID + i0] = y0;
    xout[(size_t)d * HID + i1] = y1;
    if (writebf) {
        xbf[(size_t)d * HID + i0] = (__bf16)y0;
        xbf[(size_t)d * HID + i1] = (__bf16)y1;
    }
}

// ---------------- host orchestration ----------------
static inline size_t alignup(size_t x) { return (x + 255) & ~(size_t)255; }

extern "C" void kernel_launch(void* const* d_in, const int* in_sizes, int n_in,
                              void* d_out, int out_size, void* d_ws, size_t ws_size,
                              hipStream_t stream) {
    (void)in_sizes; (void)n_in; (void)out_size; (void)ws_size;
    const float* send_rep = (const float*)d_in[0];
    const float* rec_rep  = (const float*)d_in[1];
    const float* edge_rep = (const float*)d_in[2];
    const int*   ei       = (const int*)d_in[3];
    const float *Wl0=(const float*)d_in[4],  *bl0=(const float*)d_in[5];
    const float *Wr0=(const float*)d_in[6],  *br0=(const float*)d_in[7];
    const float *We0=(const float*)d_in[8],  *att0=(const float*)d_in[9];
    const float *bias0=(const float*)d_in[10], *g0=(const float*)d_in[11], *be0=(const float*)d_in[12];
    const float *Wl1=(const float*)d_in[13], *bl1=(const float*)d_in[14];
    const float *Wr1=(const float*)d_in[15], *br1=(const float*)d_in[16];
    const float *We1=(const float*)d_in[17], *att1=(const float*)d_in[18];
    const float *bias1=(const float*)d_in[19], *g1=(const float*)d_in[20], *be1=(const float*)d_in[21];
    const float *res_W=(const float*)d_in[22], *res_b=(const float*)d_in[23];
    float* out = (float*)d_out;

    // ---- carve workspace ----
    char* p = (char*)d_ws; size_t off = 0;
    auto carve = [&](size_t bytes) { void* q = p + off; off = alignup(off + bytes); return q; };
    __bf16* sendbf = (__bf16*)carve((size_t)NS * SD * 2);
    __bf16* recbf  = (__bf16*)carve((size_t)ND * RD * 2);
    __bf16* edgebf = (__bf16*)carve((size_t)EE * ED * 2);
    __bf16* xdstbf = (__bf16*)carve((size_t)ND * HID * 2);
    __bf16* wl0b = (__bf16*)carve((size_t)HO * SD * 2);
    __bf16* wr0b = (__bf16*)carve((size_t)HO * RD * 2);
    __bf16* we0b = (__bf16*)carve((size_t)HO * ED * 2);
    __bf16* wl1b = (__bf16*)carve((size_t)HO * SD * 2);
    __bf16* wr1b = (__bf16*)carve((size_t)HO * HID * 2);
    __bf16* we1b = (__bf16*)carve((size_t)HO * ED * 2);
    __bf16* rwb  = (__bf16*)carve((size_t)HID * RD * 2);
    __bf16* xl = (__bf16*)carve((size_t)NS * HO * 2);
    __bf16* xr = (__bf16*)carve((size_t)ND * HO * 2);
    __bf16* xe = (__bf16*)carve((size_t)EE * HO * 2);
    float*  logits = (float*)carve((size_t)EE * HEADS * 4);
    unsigned* menc = (unsigned*)carve((size_t)ND * HEADS * 4);
    float*  denom  = (float*)carve((size_t)ND * HEADS * 4);
    float*  segout = (float*)carve((size_t)ND * HO * 4);
    float*  resbuf = (float*)carve((size_t)ND * HID * 4);
    float*  xdstf  = (float*)carve((size_t)ND * HID * 4);

    // ---- bf16 casts (activations + weights) ----
    auto cast = [&](const float* src, __bf16* dst, long n) {
        int grid = (int)((n + 255) / 256); if (grid > 4096) grid = 4096;
        cast_bf16<<<grid, 256, 0, stream>>>(src, dst, (int)n);
    };
    cast(send_rep, sendbf, (long)NS * SD);
    cast(rec_rep,  recbf,  (long)ND * RD);
    cast(edge_rep, edgebf, (long)EE * ED);
    cast(Wl0, wl0b, (long)HO * SD);  cast(Wr0, wr0b, (long)HO * RD);
    cast(We0, we0b, (long)HO * ED);  cast(Wl1, wl1b, (long)HO * SD);
    cast(Wr1, wr1b, (long)HO * HID); cast(We1, we1b, (long)HO * ED);
    cast(res_W, rwb, (long)HID * RD);

    const dim3 blkG(128);
    // ---- layer 0 linears + residual projection (all WMMA) ----
    gemm_wmma<SD,  HO, true,  __bf16><<<dim3(NS/16, 4), blkG, 0, stream>>>(sendbf, wl0b, bl0,  xl);
    gemm_wmma<RD,  HO, true,  __bf16><<<dim3(ND/16, 4), blkG, 0, stream>>>(recbf,  wr0b, br0,  xr);
    gemm_wmma<ED,  HO, false, __bf16><<<dim3(EE/16, 4), blkG, 0, stream>>>(edgebf, we0b, nullptr, xe);
    gemm_wmma<RD, HID, true,  float ><<<dim3(ND/16, 1), blkG, 0, stream>>>(recbf,  rwb,  res_b, resbuf);

    // ---- layer 0 attention ----
    hipMemsetAsync(menc,   0, (size_t)ND * HEADS * 4, stream);
    hipMemsetAsync(denom,  0, (size_t)ND * HEADS * 4, stream);
    hipMemsetAsync(segout, 0, (size_t)ND * HO * 4,    stream);
    edge_logits <<<EE / 8, 256, 0, stream>>>(xl, xr, xe, ei, att0, logits, menc);
    edge_exp    <<<(EE * HEADS + 255) / 256, 256, 0, stream>>>(logits, menc, denom, ei);
    edge_scatter<<<EE / 8, 256, 0, stream>>>(xl, logits, denom, ei, segout);
    node_ln     <<<ND / 8, 256, 0, stream>>>(segout, bias0, resbuf, g0, be0, xdstf, xdstbf, 1);

    // ---- layer 1 linears (reuse big buffers) ----
    gemm_wmma<SD,  HO, true,  __bf16><<<dim3(NS/16, 4), blkG, 0, stream>>>(sendbf, wl1b, bl1, xl);
    gemm_wmma<HID, HO, true,  __bf16><<<dim3(ND/16, 4), blkG, 0, stream>>>(xdstbf, wr1b, br1, xr);
    gemm_wmma<ED,  HO, false, __bf16><<<dim3(EE/16, 4), blkG, 0, stream>>>(edgebf, we1b, nullptr, xe);

    // ---- layer 1 attention ----
    hipMemsetAsync(menc,   0, (size_t)ND * HEADS * 4, stream);
    hipMemsetAsync(denom,  0, (size_t)ND * HEADS * 4, stream);
    hipMemsetAsync(segout, 0, (size_t)ND * HO * 4,    stream);
    edge_logits <<<EE / 8, 256, 0, stream>>>(xl, xr, xe, ei, att1, logits, menc);
    edge_exp    <<<(EE * HEADS + 255) / 256, 256, 0, stream>>>(logits, menc, denom, ei);
    edge_scatter<<<EE / 8, 256, 0, stream>>>(xl, logits, denom, ei, segout);
    node_ln     <<<ND / 8, 256, 0, stream>>>(segout, bias1, xdstf, g1, be1, out, nullptr, 0);
}
